// CausalSelfAttention_24257975288414
// MI455X (gfx1250) — compile-verified, matrix operand order checked by reading, not run on hardware
//
#include <hip/hip_runtime.h>
#include <hip/hip_bf16.h>
#include <hip/hip_fp16.h>

typedef _Float16 half4 __attribute__((ext_vector_type(4)));
typedef _Float16 half8 __attribute__((ext_vector_type(8)));
typedef _Float16 v16h  __attribute__((ext_vector_type(16)));
typedef float    v8f   __attribute__((ext_vector_type(8)));
typedef unsigned int       u32;
typedef unsigned long long u64;
typedef u32 v4u __attribute__((ext_vector_type(4)));
typedef int  v8i __attribute__((ext_vector_type(8)));
typedef int  v4i __attribute__((ext_vector_type(4)));

#define E_DIM 1024
#define D_DIM 128
#define S_LEN 2048
#define LOG2E 1.44269504088896340736f
#define INV_SQRT_D 0.08838834764831845f  // 1/sqrt(128)

// K tile in LDS: 32 rows x 256B data + 16B pad  -> stride 272 B (136 halves)
#define KROW_H 136
// V tile in LDS: 128 rows x 64B data + 16B pad  -> stride 80 B (40 halves)
#define VROW_H 40

#if defined(__has_builtin)
#  if __has_builtin(__builtin_amdgcn_tensor_load_to_lds)
#    define HAS_TDM 1
#  endif
#endif
#ifndef HAS_TDM
#  define HAS_TDM 0
#endif

static __device__ __forceinline__ v16h mkfrag(half8 lo, half8 hi) {
    union { v16h v; half8 h[2]; } u;
    u.h[0] = lo; u.h[1] = hi;
    return u.v;
}
static __device__ __forceinline__ half8 ldg8(const _Float16* p) {
    return *reinterpret_cast<const half8*>(p);
}

#if HAS_TDM
// Build a D# (ISA 8.3/8.4) and issue TENSOR_LOAD_TO_LDS.
//   data_size = 8B units; tile_d1 == 0 -> 1-D tile; pad_enable always on.
static __device__ __forceinline__ void tdm_load(
        u32 lds_off, const void* gptr, u32 tile_d0, u32 tile_d1,
        u64 t0_stride, u32 pad_interval, u32 pad_amount) {
    const u64 ga = (u64)gptr;
    v4u g0;
    g0[0] = 1u;                                        // count=1, user mode
    g0[1] = lds_off;                                   // lds_addr
    g0[2] = (u32)(ga & 0xFFFFFFFFull);                 // global_addr[31:0]
    g0[3] = (u32)((ga >> 32) & 0x01FFFFFFull) | (2u << 30); // addr[56:32]|type=2
    const u32 td0 = 1u << 20, td1 = 1u << 20;          // generous dims: no OOB clip
    v8i g1;
    g1[0] = (int)((3u << 16) | (1u << 20) |            // data_size=8B, pad_enable
                  (pad_interval << 22) | (pad_amount << 25));
    g1[1] = (int)((td0 & 0xFFFFu) << 16);              // tensor_dim0[15:0]
    g1[2] = (int)(((td0 >> 16) & 0xFFFFu) | ((td1 & 0xFFFFu) << 16));
    g1[3] = (int)(((td1 >> 16) & 0xFFFFu) | (tile_d0 << 16));
    g1[4] = (int)(tile_d1 & 0xFFFFu);                  // tile_dim1 (tile_dim2=0)
    g1[5] = (int)(t0_stride & 0xFFFFFFFFull);          // tensor_dim0_stride[31:0]
    g1[6] = (int)((t0_stride >> 32) & 0xFFFFull);      // stride[47:32]
    g1[7] = 0;
    v4i z4 = {0, 0, 0, 0};
#if defined(__clang_major__) && (__clang_major__ >= 23)
    v8i z8 = {0, 0, 0, 0, 0, 0, 0, 0};
    __builtin_amdgcn_tensor_load_to_lds(g0, g1, z4, z4, z8, 0);
#else
    __builtin_amdgcn_tensor_load_to_lds(g0, g1, z4, z4, 0);
#endif
}
#endif

// ---------------------------------------------------------------------------
// Kernel 2 (defined first so the TDM sequence leads the disassembly):
// causal flash attention with TDM-staged, double-buffered K/V in LDS.
//   grid (16, 4), block 256 (8 waves). Wave w owns query tile (bx*8+w).
//   Wave 0 drives TENSOR_LOAD_TO_LDS (TENSORcnt); block barriers publish tiles.
//   Softmax denominator rides the matrix pipe: o9 = wmma(P, ones, o9).
// ---------------------------------------------------------------------------
__global__ __launch_bounds__(256) void flash_kernel(
        const _Float16* __restrict__ Qh, const _Float16* __restrict__ Kh,
        const _Float16* __restrict__ Vt, float* __restrict__ out) {
    __shared__ _Float16 sK[2][32 * KROW_H];        // 2 x 8.5 KB (TDM-padded rows)
    __shared__ _Float16 sV[2][128 * VROW_H];       // 2 x 10 KB (TDM-padded rows)
    __shared__ _Float16 sP[8 * 640];               // P bounce, stride 40 halves

    const int tid  = threadIdx.x;
    const int wave = tid >> 5, lane = tid & 31;
    const int ln = lane & 15, g = lane >> 4;
    const int bx = blockIdx.x;
    const int b  = blockIdx.y;
    const int qbase = (bx * 8 + wave) * 16;

    const _Float16* Qb = Qh + (size_t)(b * S_LEN) * D_DIM;
    const _Float16* Kb = Kh + (size_t)(b * S_LEN) * D_DIM;
    const _Float16* Vb = Vt + (size_t)b * D_DIM * S_LEN;

    // Q fragments: 16x128 (4 k-steps of 32), resident in VGPRs.
    v16h qf[4];
    #pragma unroll
    for (int ds = 0; ds < 4; ++ds) {
        const _Float16* qp = Qb + (size_t)(qbase + ln) * D_DIM + ds * 32 + g * 8;
        qf[ds] = mkfrag(ldg8(qp), ldg8(qp + 16));
    }

    v16h ones;
    #pragma unroll
    for (int i = 0; i < 16; ++i) ones[i] = (_Float16)1.0f;

    v8f o[8];
    #pragma unroll
    for (int i = 0; i < 8; ++i) o[i] = (v8f){};
    v8f o9 = {};                                   // running softmax denominator
    float mrow[8];
    #pragma unroll
    for (int r = 0; r < 8; ++r) mrow[r] = -3.0e38f;

    _Float16* pbuf = &sP[wave * 640];
    const int nkb     = (qbase >> 5) + 1;          // this wave's causal trip count
    const int nkb_blk = 4 * bx + 4;                // block-max trip count

#if HAS_TDM
    const u32 offK[2] = { (u32)(size_t)&sK[0][0], (u32)(size_t)&sK[1][0] };
    const u32 offV[2] = { (u32)(size_t)&sV[0][0], (u32)(size_t)&sV[1][0] };
    if (wave == 0) {
        // K tile: 8192B contiguous, 1-D; pad 4 dwords every 64 dwords (272B rows).
        tdm_load(offK[0], Kb, 1024, 0, 1024, 5, 3);
        // V tile: 128 rows x 64B, row stride 4096B; pad 4 dw every 16 dw (80B rows).
        tdm_load(offV[0], Vb, 8, 128, 512, 3, 3);
    }
#else
    for (int i = tid; i < 512; i += 256) {
        int lb = i * 16, row = lb >> 8, w = lb & 255;
        *(uint4*)((char*)&sK[0][0] + row * 272 + w) = *((const uint4*)Kb + i);
        int rv = lb >> 6, wv = lb & 63;
        *(uint4*)((char*)&sV[0][0] + rv * 80 + wv) =
            *(const uint4*)((const char*)(Vb + (size_t)rv * S_LEN) + wv);
    }
#endif

    for (int kb = 0; kb < nkb_blk; ++kb) {
        const int buf   = kb & 1;
        const int tbase = kb * 32;
        // Issue DMA for the next key block into the other buffer.
        if (kb + 1 < nkb_blk) {
            const int tn = tbase + 32;
#if HAS_TDM
            if (wave == 0) {
                tdm_load(offK[buf ^ 1], Kb + (size_t)tn * D_DIM, 1024, 0, 1024, 5, 3);
                tdm_load(offV[buf ^ 1], Vb + tn, 8, 128, 512, 3, 3);
                __builtin_amdgcn_s_wait_tensorcnt(2);  // current buffer complete
            }
#else
            for (int i = tid; i < 512; i += 256) {
                int lb = i * 16, row = lb >> 8, w = lb & 255;
                *(uint4*)((char*)&sK[buf ^ 1][0] + row * 272 + w) =
                    *((const uint4*)(Kb + (size_t)tn * D_DIM) + i);
                int rv = lb >> 6, wv = lb & 63;
                *(uint4*)((char*)&sV[buf ^ 1][0] + rv * 80 + wv) =
                    *(const uint4*)((const char*)(Vb + (size_t)rv * S_LEN + tn) + wv);
            }
#endif
        }
#if HAS_TDM
        else if (wave == 0) {
            __builtin_amdgcn_s_wait_tensorcnt(0);
        }
#endif
        __syncthreads();                            // publish current buffer

        if (kb < nkb) {
            // S = Q (16x128) @ K^T (128x32) -> two 16x16 tiles, 8 WMMAs from LDS.
            v8f s0 = {}, s1 = {};
            #pragma unroll
            for (int ds = 0; ds < 4; ++ds) {
                const _Float16* kp0 = &sK[buf][(size_t)ln * KROW_H + ds * 32 + g * 8];
                v16h k0 = mkfrag(*(const half8*)kp0, *(const half8*)(kp0 + 16));
                s0 = __builtin_amdgcn_wmma_f32_16x16x32_f16(
                         false, qf[ds], false, k0, (short)0, s0, false, false);
                const _Float16* kp1 = kp0 + 16 * KROW_H;
                v16h k1 = mkfrag(*(const half8*)kp1, *(const half8*)(kp1 + 16));
                s1 = __builtin_amdgcn_wmma_f32_16x16x32_f16(
                         false, qf[ds], false, k1, (short)0, s1, false, false);
            }
            const bool last = (kb == nkb - 1);
            // Online softmax: row max via width-16 shuffles; denominator via o9.
            #pragma unroll
            for (int r = 0; r < 8; ++r) {
                float v0 = s0[r] * LOG2E;          // Q carries 1/sqrt(D)
                float v1 = s1[r] * LOG2E;
                if (last) {
                    const int srow = qbase + r + 8 * g;
                    if (tbase + ln      > srow) v0 = -3.0e38f;
                    if (tbase + 16 + ln > srow) v1 = -3.0e38f;
                }
                float rmax = fmaxf(v0, v1);
                rmax = fmaxf(rmax, __shfl_xor(rmax, 1, 16));
                rmax = fmaxf(rmax, __shfl_xor(rmax, 2, 16));
                rmax = fmaxf(rmax, __shfl_xor(rmax, 4, 16));
                rmax = fmaxf(rmax, __shfl_xor(rmax, 8, 16));
                const float mnew  = fmaxf(mrow[r], rmax);
                const float alpha = exp2f(mrow[r] - mnew);
                mrow[r] = mnew;
                const float p0 = exp2f(v0 - mnew);
                const float p1 = exp2f(v1 - mnew);
                #pragma unroll
                for (int dn = 0; dn < 8; ++dn) o[dn][r] *= alpha;
                o9[r] *= alpha;
                const int m = r + 8 * g;           // bounce P to LDS (C -> A layout)
                pbuf[m * 40 + ln]      = (_Float16)p0;
                pbuf[m * 40 + 16 + ln] = (_Float16)p1;
            }
            asm volatile("s_wait_dscnt 0" ::: "memory");  // per-wave LDS RAW fence
            const _Float16* pp = pbuf + ln * 40 + g * 8;
            v16h pf = mkfrag(*(const half8*)pp, *(const half8*)(pp + 16));
            // O += P (16x32) @ V (32x128): 8 WMMAs, B-frags from padded LDS.
            #pragma unroll
            for (int dn = 0; dn < 8; ++dn) {
                const _Float16* vp = &sV[buf][(size_t)(dn * 16 + ln) * VROW_H + g * 8];
                v16h vf = mkfrag(*(const half8*)vp, *(const half8*)(vp + 16));
                o[dn] = __builtin_amdgcn_wmma_f32_16x16x32_f16(
                            false, pf, false, vf, (short)0, o[dn], false, false);
            }
            // Row-sum of P on the matrix pipe (replaces shuffle-chain reduction).
            o9 = __builtin_amdgcn_wmma_f32_16x16x32_f16(
                     false, pf, false, ones, (short)0, o9, false, false);
        }
        __syncthreads();                            // buffer free for next DMA
    }

    // Epilogue: normalize by o9 (rowsum, replicated per lane group), store f32.
    float inv[8];
    #pragma unroll
    for (int r = 0; r < 8; ++r) inv[r] = 1.0f / o9[r];
    float* ob = out + (size_t)(b * S_LEN + qbase) * D_DIM;
    #pragma unroll
    for (int dn = 0; dn < 8; ++dn) {
        #pragma unroll
        for (int r = 0; r < 8; ++r) {
            const int m = r + 8 * g;
            ob[(size_t)m * D_DIM + dn * 16 + ln] = o[dn][r] * inv[r];
        }
    }
}

// ---------------------------------------------------------------------------
// Kernel 0: transpose + f32->f16 convert weights: W[1024][128] -> WT[mat][128][1024]
// ---------------------------------------------------------------------------
__global__ __launch_bounds__(256) void wt_kernel(const float* __restrict__ Wq,
                                                 const float* __restrict__ Wk,
                                                 const float* __restrict__ Wv,
                                                 _Float16* __restrict__ WT) {
    int idx = blockIdx.x * 256 + threadIdx.x;      // [0, 3*131072)
    int mat = idx >> 17;
    int rem = idx & 131071;
    int d   = rem >> 10;
    int e   = rem & 1023;
    const float* W = (mat == 0) ? Wq : ((mat == 1) ? Wk : Wv);
    WT[idx] = (_Float16)W[e * D_DIM + d];          // coalesced write along e
}

// ---------------------------------------------------------------------------
// Kernel 1: fused QKV projection (WMMA f16 -> f32 accum)
// ---------------------------------------------------------------------------
__global__ __launch_bounds__(256) void qkv_kernel(
        const float* __restrict__ x, const _Float16* __restrict__ WT,
        const float* __restrict__ bq, const float* __restrict__ bk,
        const float* __restrict__ bv,
        _Float16* __restrict__ Qh, _Float16* __restrict__ Kh,
        _Float16* __restrict__ Vt) {
    __shared__ _Float16 sA[16 * 1032];             // padded stride vs bank conflicts
    const int m0  = blockIdx.x * 16;
    const int tid = threadIdx.x;

    #pragma unroll 4
    for (int i = 0; i < 16; ++i) {
        int idx4 = tid + i * 256;                  // float4 index, 4096 total
        int row  = idx4 >> 8;
        int c4   = (idx4 & 255) * 4;
        float4 f = reinterpret_cast<const float4*>(
                       x + (size_t)(m0 + row) * E_DIM)[idx4 & 255];
        half4 h;
        h[0] = (_Float16)f.x; h[1] = (_Float16)f.y;
        h[2] = (_Float16)f.z; h[3] = (_Float16)f.w;
        *reinterpret_cast<half4*>(&sA[row * 1032 + c4]) = h;
    }
    __syncthreads();

    const int wave = tid >> 5, lane = tid & 31;
    const int ln = lane & 15, g = lane >> 4;
    const int ntile = blockIdx.y * 8 + wave;       // 0..23
    const int mat   = ntile >> 3;                  // 0=q 1=k 2=v
    const int ncol  = (ntile & 7) * 16;

    const _Float16* Wp = WT + (size_t)mat * 131072 + (size_t)(ncol + ln) * E_DIM;
    v8f acc = {};
    for (int kb = 0; kb < E_DIM; kb += 32) {
        const _Float16* ap = &sA[ln * 1032 + kb + g * 8];
        v16h a = mkfrag(*(const half8*)ap, *(const half8*)(ap + 16));
        const _Float16* bp = Wp + kb + g * 8;
        v16h bfr = mkfrag(ldg8(bp), ldg8(bp + 16));
        acc = __builtin_amdgcn_wmma_f32_16x16x32_f16(
                  false, a, false, bfr, (short)0, acc, false, false);
    }
    const float* bias = (mat == 0) ? bq : ((mat == 1) ? bk : bv);
    const float bval  = bias[ncol + ln];
    const float scal  = (mat == 0) ? INV_SQRT_D : 1.0f;

    #pragma unroll
    for (int r = 0; r < 8; ++r) {
        float v  = (acc[r] + bval) * scal;
        int   m  = r + 8 * g;
        int   sr = m0 + m;                         // flattened B*S row
        _Float16 hv = (_Float16)v;
        if (mat == 0) {
            Qh[(size_t)sr * D_DIM + ncol + ln] = hv;
        } else if (mat == 1) {
            Kh[(size_t)sr * D_DIM + ncol + ln] = hv;
        } else {
            int b = sr >> 11, t = sr & (S_LEN - 1);
            Vt[(size_t)b * D_DIM * S_LEN + (size_t)(ncol + ln) * S_LEN + t] = hv;
        }
    }
}

// ---------------------------------------------------------------------------
extern "C" void kernel_launch(void* const* d_in, const int* in_sizes, int n_in,
                              void* d_out, int out_size, void* d_ws, size_t ws_size,
                              hipStream_t stream) {
    const float* x  = (const float*)d_in[0];
    const float* Wq = (const float*)d_in[1];
    const float* bq = (const float*)d_in[2];
    const float* Wk = (const float*)d_in[3];
    const float* bk = (const float*)d_in[4];
    const float* Wv = (const float*)d_in[5];
    const float* bv = (const float*)d_in[6];
    float* out = (float*)d_out;

    char* ws = (char*)d_ws;
    _Float16* WT = (_Float16*)ws;                           // 3*128*1024 f16 = 768 KB
    _Float16* Qh = (_Float16*)(ws + 786432);                // 8192*128 f16 = 2 MB
    _Float16* Kh = (_Float16*)(ws + 786432 + 2097152);      // 2 MB
    _Float16* Vt = (_Float16*)(ws + 786432 + 2 * 2097152);  // 2 MB (transposed)

    wt_kernel<<<1536, 256, 0, stream>>>(Wq, Wk, Wv, WT);
    qkv_kernel<<<dim3(512, 3), 256, 0, stream>>>(x, WT, bq, bk, bv, Qh, Kh, Vt);
    flash_kernel<<<dim3(16, 4), 256, 0, stream>>>(Qh, Kh, Vt, out);
}
